// MLA_8220567405326
// MI455X (gfx1250) — matrix operand order repaired and checked by reading, not compiled, not run on hardware
//
#include <hip/hip_runtime.h>
#include <hip/hip_bf16.h>

// ---------------------------------------------------------------------------
// MLA forward for gfx1250 (MI455X). All matmuls on v_wmma_f32_16x16x32_bf16,
// fp32 accumulation, fp32 softmax. Compute-bound (~100 GFLOP vs <10us of HBM
// traffic at 23.3 TB/s), so everything routes through the WMMA pipe.
// Big GEMMs: 128x128 workgroup tile, 8 waves, double-buffered LDS with
// ASYNC global->LDS staging of the A tile (global_load_async_to_lds_b128 /
// s_wait_asynccnt) + L2 prefetch (global_prefetch_b8).
// Attention: flash, 16-query x 32-key steps, online fp32 softmax.
// ---------------------------------------------------------------------------

typedef __bf16 bf16_t;
typedef __attribute__((ext_vector_type(8)))  __bf16 v8bf;
typedef __attribute__((ext_vector_type(16))) __bf16 v16bf;
typedef __attribute__((ext_vector_type(8)))  float  v8f;

#define S_    2048
#define DIM_  2048
#define H_    16
#define QL_   1536
#define KVL_  512
#define NOPE_ 128
#define ROPE_ 64
#define VH_   128
#define QKD_  192           // NOPE + ROPE
#define KVD_  576           // KVL + ROPE
#define HKV_  256           // NOPE + VH
#define LDQ_  (H_*QKD_)     // 3072
#define LDKV_ (H_*HKV_)     // 4096
#define LDO_  (H_*VH_)      // 2048
#define LOG1E4 9.210340371976184f

static __device__ __forceinline__ v16bf cat8(v8bf lo, v8bf hi) {
  return __builtin_shufflevector(lo, hi, 0,1,2,3,4,5,6,7,8,9,10,11,12,13,14,15);
}
static __device__ __forceinline__ v8f wmma_bf(v16bf a, v16bf b, v8f c) {
  return __builtin_amdgcn_wmma_f32_16x16x32_bf16(false, a, false, b, (short)0,
                                                 c, false, false);
}

// --- CDNA5 async global->LDS copy (ASYNCcnt-tracked), 16B per lane ---------
// LDS byte offset = low 32 bits of the flat shared-memory address.
static __device__ __forceinline__ void async_ld_b128(const bf16_t* g, bf16_t* l) {
  unsigned lds = (unsigned)(uintptr_t)l;
  unsigned long long ga = (unsigned long long)(uintptr_t)g;
  asm volatile("global_load_async_to_lds_b128 %0, %1, off"
               :: "v"(lds), "v"(ga) : "memory");
}
static __device__ __forceinline__ void wait_async() {
  asm volatile("s_wait_asynccnt 0x0" ::: "memory");
}

// ------------------------- fp32 -> bf16 convert ----------------------------
__global__ void f2bf_kernel(const float* __restrict__ in, bf16_t* __restrict__ out,
                            size_t n) {
  size_t i = (size_t)blockIdx.x * blockDim.x + threadIdx.x;
  if (i < n) out[i] = (bf16_t)in[i];
}

// ------------------------- RMS norm -> bf16 --------------------------------
__global__ void rms_bf16_kernel(const float* __restrict__ in,
                                const float* __restrict__ w,
                                bf16_t* __restrict__ out, int L, int pitch) {
  int r = blockIdx.x;
  const float* row = in + (size_t)r * pitch;
  __shared__ float red[256];
  float ss = 0.f;
  for (int i = threadIdx.x; i < L; i += 256) { float v = row[i]; ss += v * v; }
  red[threadIdx.x] = ss;
  __syncthreads();
  for (int s = 128; s > 0; s >>= 1) {
    if ((int)threadIdx.x < s) red[threadIdx.x] += red[threadIdx.x + s];
    __syncthreads();
  }
  float scale = rsqrtf(red[0] / (float)L + 1e-6f);
  for (int i = threadIdx.x; i < L; i += 256)
    out[(size_t)r * L + i] = (bf16_t)(row[i] * scale * w[i]);
}

// -------------------- q post: copy nope + rope pe -> bf16 ------------------
__global__ __launch_bounds__(128) void q_post_kernel(const float* __restrict__ qf,
                                                     bf16_t* __restrict__ qb) {
  int s = blockIdx.x, h = blockIdx.y, t = threadIdx.x;
  const float* src = qf + ((size_t)s * H_ + h) * QKD_;
  bf16_t*      dst = qb + ((size_t)s * H_ + h) * QKD_;
  dst[t] = (bf16_t)src[t];                     // nope, d = 0..127
  if (t < ROPE_ / 2) {
    float invf = __expf(-(2.f * t / ROPE_) * LOG1E4);
    float ang = (float)s * invf, sn, cs;
    sincosf(ang, &sn, &cs);
    float x0 = src[NOPE_ + 2 * t], x1 = src[NOPE_ + 2 * t + 1];
    dst[NOPE_ + 2 * t]     = (bf16_t)(x0 * cs - x1 * sn);
    dst[NOPE_ + 2 * t + 1] = (bf16_t)(x0 * sn + x1 * cs);
  }
}

// -------------------- k_pe post: rope -> bf16 ------------------------------
__global__ __launch_bounds__(32) void kpe_post_kernel(const float* __restrict__ kvf,
                                                      bf16_t* __restrict__ kpe) {
  int s = blockIdx.x, t = threadIdx.x;     // t = 0..31 (rope pairs)
  const float* src = kvf + (size_t)s * KVD_ + KVL_;
  float invf = __expf(-(2.f * t / ROPE_) * LOG1E4);
  float ang = (float)s * invf, sn, cs;
  sincosf(ang, &sn, &cs);
  float x0 = src[2 * t], x1 = src[2 * t + 1];
  kpe[(size_t)s * ROPE_ + 2 * t]     = (bf16_t)(x0 * cs - x1 * sn);
  kpe[(size_t)s * ROPE_ + 2 * t + 1] = (bf16_t)(x0 * sn + x1 * cs);
}

// ---------------- big TN GEMM: C = A(MxK) * B(NxK)^T, 128x128 tile ---------
// 256 threads (8 waves); wave = 32(M) x 64(N): 2 A-frags x 4 B-frags ->
// 8 WMMAs per K-step. Double-buffered LDS; A tile staged with async
// global->LDS copies, B tile via load + transpose-store; L2 prefetch.
#define BM 128
#define BN 128
#define BK 32
__global__ __launch_bounds__(256) void gemm_bf16_tn_big(const bf16_t* __restrict__ A,
                                                        const bf16_t* __restrict__ B,
                                                        float* __restrict__ C,
                                                        int M, int N, int K) {
  __shared__ __attribute__((aligned(32))) bf16_t sA[2][BM * BK];  // [m][k]
  __shared__ __attribute__((aligned(32))) bf16_t sB[2][BK * BN];  // [k][n]
  const int tid = threadIdx.x;
  const int lane = tid & 31, wave = tid >> 5;
  const int wm = (wave & 3) * 32, wn = (wave >> 2) * 64;
  const int mBase = blockIdx.y * BM, nBase = blockIdx.x * BN;

  v8f acc[2][4] = {};

  auto stage = [&](int k0, int buf) {
#pragma unroll
    for (int i = 0; i < 2; ++i) {   // A: 512 x 16B chunks, direct to LDS
      int idx = tid + i * 256;
      int row = idx >> 2, col = (idx & 3) * 8;
      async_ld_b128(A + (size_t)(mBase + row) * K + k0 + col,
                    &sA[buf][row * BK + col]);
    }
#pragma unroll
    for (int i = 0; i < 2; ++i) {   // B -> [k][n] transpose via VGPRs
      int idx = tid + i * 256;
      int n = idx >> 2, ck = (idx & 3) * 8;
      v8bf v = *(const v8bf*)(B + (size_t)(nBase + n) * K + k0 + ck);
#pragma unroll
      for (int j = 0; j < 8; ++j) sB[buf][(ck + j) * BN + n] = v[j];
    }
  };

  stage(0, 0);
  wait_async();
  __syncthreads();

  const int arow = lane & 15, koff = (lane >> 4) * 8;
  for (int k0 = 0; k0 < K; k0 += BK) {
    int buf = (k0 / BK) & 1;
    if (k0 + BK < K) {
      if (k0 + 2 * BK < K) {   // pull K+2 tile toward L2 (global_prefetch_b8)
        __builtin_prefetch(A + (size_t)(mBase + (tid >> 1)) * K + k0 + 2 * BK +
                               (tid & 1) * 16, 0, 1);
        __builtin_prefetch(B + (size_t)(nBase + (tid >> 1)) * K + k0 + 2 * BK +
                               (tid & 1) * 16, 0, 1);
      }
      stage(k0 + BK, buf ^ 1);   // overlaps with compute below (other buffer)
    }
    v16bf a0 = cat8(*(const v8bf*)&sA[buf][(wm + arow) * BK + koff],
                    *(const v8bf*)&sA[buf][(wm + arow) * BK + koff + 16]);
    v16bf a1 = cat8(*(const v8bf*)&sA[buf][(wm + 16 + arow) * BK + koff],
                    *(const v8bf*)&sA[buf][(wm + 16 + arow) * BK + koff + 16]);
#pragma unroll
    for (int j = 0; j < 4; ++j) {
      v16bf b;
#pragma unroll
      for (int s = 0; s < 16; ++s)
        b[s] = sB[buf][(koff + (s & 7) + (s >> 3) * 16) * BN + wn + j * 16 + arow];
      acc[0][j] = wmma_bf(a0, b, acc[0][j]);
      acc[1][j] = wmma_bf(a1, b, acc[1][j]);
    }
    wait_async();       // next tile's async LDS writes done before barrier
    __syncthreads();
  }

  const int rOff = (lane >> 4) * 8, colL = lane & 15;
#pragma unroll
  for (int half = 0; half < 2; ++half)
#pragma unroll
    for (int j = 0; j < 4; ++j)
#pragma unroll
      for (int e = 0; e < 8; ++e)
        C[(size_t)(mBase + wm + half * 16 + e + rOff) * N +
          nBase + wn + j * 16 + colL] = acc[half][j][e];
}

// ---------------- small TN GEMM (64x64 tile) for N not /128 ----------------
__global__ __launch_bounds__(128) void gemm_bf16_tn(const bf16_t* __restrict__ A,
                                                    const bf16_t* __restrict__ B,
                                                    float* __restrict__ C,
                                                    int M, int N, int K) {
  __shared__ __attribute__((aligned(32))) bf16_t sA[64 * 32];
  __shared__ __attribute__((aligned(32))) bf16_t sB[32 * 64];
  const int tid = threadIdx.x;
  const int lane = tid & 31, wave = tid >> 5;
  const int mBase = blockIdx.y * 64, nBase = blockIdx.x * 64;

  v8f acc[4] = {};
  const int arow = lane & 15, koff = (lane >> 4) * 8;

  for (int k0 = 0; k0 < 0 + K; k0 += 32) {
#pragma unroll
    for (int i = 0; i < 2; ++i) {
      int idx = tid + i * 128;
      int row = idx >> 2, col = (idx & 3) * 8;
      async_ld_b128(A + (size_t)(mBase + row) * K + k0 + col,
                    &sA[row * 32 + col]);
    }
#pragma unroll
    for (int i = 0; i < 2; ++i) {
      int idx = tid + i * 128;
      int n = idx >> 2, ck = (idx & 3) * 8;
      v8bf v = *(const v8bf*)(B + (size_t)(nBase + n) * K + k0 + ck);
#pragma unroll
      for (int j = 0; j < 8; ++j) sB[(ck + j) * 64 + n] = v[j];
    }
    wait_async();
    __syncthreads();

    v16bf a = cat8(*(const v8bf*)&sA[(wave * 16 + arow) * 32 + koff],
                   *(const v8bf*)&sA[(wave * 16 + arow) * 32 + koff + 16]);
#pragma unroll
    for (int t = 0; t < 4; ++t) {
      v16bf b;
#pragma unroll
      for (int s = 0; s < 16; ++s)
        b[s] = sB[(koff + (s & 7) + (s >> 3) * 16) * 64 + t * 16 + arow];
      acc[t] = wmma_bf(a, b, acc[t]);
    }
    __syncthreads();
  }

  const int rOff = (lane >> 4) * 8, colL = lane & 15;
#pragma unroll
  for (int t = 0; t < 4; ++t)
#pragma unroll
    for (int e = 0; e < 8; ++e)
      C[(size_t)(mBase + wave * 16 + e + rOff) * N + nBase + t * 16 + colL] =
          acc[t][e];
}

// -------------------- flash attention (per head, 16-query tiles) -----------
__global__ __launch_bounds__(32) void mla_attn_kernel(const bf16_t* __restrict__ qb,
                                                      const bf16_t* __restrict__ kvu,
                                                      const bf16_t* __restrict__ kpe,
                                                      float* __restrict__ out) {
  const int q0 = blockIdx.x * 16;
  const int h  = blockIdx.y;
  const int lane = threadIdx.x;

  __shared__ __attribute__((aligned(32))) bf16_t sQ[16 * QKD_];   // [m][d]
  __shared__ __attribute__((aligned(32))) bf16_t sKT[QKD_ * 32];  // [d][key]
  __shared__ __attribute__((aligned(32))) bf16_t sP[16 * 32];     // [m][key]

  // stage 16xQKD query tile straight into LDS (async copies)
  for (int idx = lane; idx < 16 * (QKD_ / 8); idx += 32) {
    int m = idx / (QKD_ / 8), d0 = (idx % (QKD_ / 8)) * 8;
    async_ld_b128(qb + (size_t)(q0 + m) * LDQ_ + h * QKD_ + d0,
                  &sQ[m * QKD_ + d0]);
  }
  wait_async();

  v8f acc[8] = {};
  float mi[8], li[8];
#pragma unroll
  for (int e = 0; e < 8; ++e) { mi[e] = -1e30f; li[e] = 0.f; }

  const float scale = rsqrtf((float)QKD_);
  const int rOff = (lane >> 4) * 8, col = lane & 15;
  const int qrow = lane & 15, koff = (lane >> 4) * 8;

  for (int t0 = 0; t0 < q0 + 16; t0 += 32) {
    __syncthreads();
    // stage K^T (32 keys x 192 d -> [d][key]); d<128 from k_nope, else k_pe
    for (int idx = lane; idx < 32 * (QKD_ / 8); idx += 32) {
      int key = idx / (QKD_ / 8), d0 = (idx % (QKD_ / 8)) * 8;
      v8bf v;
      if (d0 < NOPE_)
        v = *(const v8bf*)(kvu + (size_t)(t0 + key) * LDKV_ + h * HKV_ + d0);
      else
        v = *(const v8bf*)(kpe + (size_t)(t0 + key) * ROPE_ + (d0 - NOPE_));
#pragma unroll
      for (int j = 0; j < 8; ++j) sKT[(d0 + j) * 32 + key] = v[j];
    }
    __syncthreads();

    v8f s0 = {}, s1 = {};
#pragma unroll
    for (int kk = 0; kk < QKD_ / 32; ++kk) {
      int dd = kk * 32;
      v16bf a = cat8(*(const v8bf*)&sQ[qrow * QKD_ + dd + koff],
                     *(const v8bf*)&sQ[qrow * QKD_ + dd + koff + 16]);
      v16bf b0, b1;
#pragma unroll
      for (int s = 0; s < 16; ++s) {
        int kr = dd + koff + (s & 7) + (s >> 3) * 16;
        b0[s] = sKT[kr * 32 + 0 + qrow];
        b1[s] = sKT[kr * 32 + 16 + qrow];
      }
      s0 = wmma_bf(a, b0, s0);
      s1 = wmma_bf(a, b1, s1);
    }

    // online softmax (row stats shared across the 16 lanes of each half-wave)
#pragma unroll
    for (int e = 0; e < 8; ++e) {
      int qi = q0 + e + rOff;
      float v0 = s0[e] * scale, v1 = s1[e] * scale;
      if (t0 + col > qi)      v0 = -1e30f;
      if (t0 + 16 + col > qi) v1 = -1e30f;
      float mt = fmaxf(v0, v1);
#pragma unroll
      for (int off = 1; off < 16; off <<= 1)
        mt = fmaxf(mt, __shfl_xor(mt, off, 32));
      float mnew  = fmaxf(mi[e], mt);
      float alpha = __expf(mi[e] - mnew);
      float p0 = __expf(v0 - mnew), p1 = __expf(v1 - mnew);
      float rs = p0 + p1;
#pragma unroll
      for (int off = 1; off < 16; off <<= 1)
        rs += __shfl_xor(rs, off, 32);
      li[e] = li[e] * alpha + rs;
      mi[e] = mnew;
#pragma unroll
      for (int t = 0; t < 8; ++t) acc[t][e] *= alpha;
      int m = e + rOff;
      sP[m * 32 + col]      = (bf16_t)p0;
      sP[m * 32 + 16 + col] = (bf16_t)p1;
    }
    __syncthreads();

    // O += P(16x32) * V(32x16 per tile), 8 V tiles cover VH=128
    v16bf p = cat8(*(const v8bf*)&sP[qrow * 32 + koff],
                   *(const v8bf*)&sP[qrow * 32 + koff + 16]);
#pragma unroll
    for (int t = 0; t < 8; ++t) {
      v16bf vb;
#pragma unroll
      for (int s = 0; s < 16; ++s)
        vb[s] = kvu[(size_t)(t0 + koff + (s & 7) + (s >> 3) * 16) * LDKV_ +
                    h * HKV_ + NOPE_ + t * 16 + col];
      acc[t] = wmma_bf(p, vb, acc[t]);
    }
  }

#pragma unroll
  for (int t = 0; t < 8; ++t)
#pragma unroll
    for (int e = 0; e < 8; ++e)
      out[(size_t)(q0 + e + rOff) * LDO_ + h * VH_ + t * 16 + col] =
          acc[t][e] / li[e];
}

// ---------------------------------------------------------------------------
extern "C" void kernel_launch(void* const* d_in, const int* in_sizes, int n_in,
                              void* d_out, int out_size, void* d_ws, size_t ws_size,
                              hipStream_t stream) {
  (void)in_sizes; (void)n_in; (void)out_size; (void)ws_size;
  const float* x        = (const float*)d_in[0];
  const float* wq_down  = (const float*)d_in[1];
  const float* q_norm_w = (const float*)d_in[2];
  const float* wq_up    = (const float*)d_in[3];
  const float* wkv_down = (const float*)d_in[4];
  const float* kv_norm_w= (const float*)d_in[5];
  const float* wkv_up   = (const float*)d_in[6];
  const float* wo       = (const float*)d_in[7];
  float* out = (float*)d_out;

  char* p = (char*)d_ws;
  auto take = [&](size_t elems, size_t esz) -> void* {
    void* r = (void*)p;
    p += (elems * esz + 255) & ~(size_t)255;
    return r;
  };

  bf16_t* xb   = (bf16_t*)take((size_t)S_ * DIM_, 2);
  bf16_t* wqdb = (bf16_t*)take((size_t)QL_ * DIM_, 2);
  bf16_t* wqub = (bf16_t*)take((size_t)LDQ_ * QL_, 2);
  bf16_t* wkdb = (bf16_t*)take((size_t)KVD_ * DIM_, 2);
  bf16_t* wkub = (bf16_t*)take((size_t)LDKV_ * KVL_, 2);
  bf16_t* wob  = (bf16_t*)take((size_t)DIM_ * LDO_, 2);
  bf16_t* qnb  = (bf16_t*)take((size_t)S_ * QL_, 2);
  bf16_t* qab  = (bf16_t*)take((size_t)S_ * LDQ_, 2);
  bf16_t* kvnb = (bf16_t*)take((size_t)S_ * KVL_, 2);
  bf16_t* kpeb = (bf16_t*)take((size_t)S_ * ROPE_, 2);
  bf16_t* kvub = (bf16_t*)take((size_t)S_ * LDKV_, 2);
  bf16_t* atnb = (bf16_t*)take((size_t)S_ * LDO_, 2);
  float* qdf   = (float*)take((size_t)S_ * QL_, 4);
  float* kvf   = (float*)take((size_t)S_ * KVD_, 4);
  float* qf    = (float*)take((size_t)S_ * LDQ_, 4);
  float* kvuf  = (float*)take((size_t)S_ * LDKV_, 4);
  float* atnf  = (float*)take((size_t)S_ * LDO_, 4);

  auto cvt = [&](const float* src, bf16_t* dst, size_t n) {
    f2bf_kernel<<<dim3((unsigned)((n + 255) / 256)), dim3(256), 0, stream>>>(src, dst, n);
  };
  auto gemm_big = [&](const bf16_t* A, const bf16_t* B, float* C, int M, int N, int K) {
    gemm_bf16_tn_big<<<dim3(N / 128, M / 128), dim3(256), 0, stream>>>(A, B, C, M, N, K);
  };
  auto gemm_sm = [&](const bf16_t* A, const bf16_t* B, float* C, int M, int N, int K) {
    gemm_bf16_tn<<<dim3(N / 64, M / 64), dim3(128), 0, stream>>>(A, B, C, M, N, K);
  };

  // 0) precision conversion (one-time; weights + activations)
  cvt(x, xb, (size_t)S_ * DIM_);
  cvt(wq_down, wqdb, (size_t)QL_ * DIM_);
  cvt(wq_up, wqub, (size_t)LDQ_ * QL_);
  cvt(wkv_down, wkdb, (size_t)KVD_ * DIM_);
  cvt(wkv_up, wkub, (size_t)LDKV_ * KVL_);
  cvt(wo, wob, (size_t)DIM_ * LDO_);

  // 1) down projections
  gemm_big(xb, wqdb, qdf, S_, QL_, DIM_);      // (2048x1536)
  gemm_sm (xb, wkdb, kvf, S_, KVD_, DIM_);     // (2048x576), N not /128

  // 2) RMS norms + k_pe rope
  rms_bf16_kernel<<<dim3(S_), dim3(256), 0, stream>>>(qdf, q_norm_w, qnb, QL_, QL_);
  rms_bf16_kernel<<<dim3(S_), dim3(256), 0, stream>>>(kvf, kv_norm_w, kvnb, KVL_, KVD_);
  kpe_post_kernel<<<dim3(S_), dim3(32), 0, stream>>>(kvf, kpeb);

  // 3) up projections
  gemm_big(qnb, wqub, qf, S_, LDQ_, QL_);      // (2048x3072)
  q_post_kernel<<<dim3(S_, H_), dim3(128), 0, stream>>>(qf, qab);
  gemm_big(kvnb, wkub, kvuf, S_, LDKV_, KVL_); // (2048x4096)
  cvt(kvuf, kvub, (size_t)S_ * LDKV_);

  // 4) attention (flash, causal)
  mla_attn_kernel<<<dim3(S_ / 16, H_), dim3(32), 0, stream>>>(qab, kvub, kpeb, atnf);

  // 5) output projection
  cvt(atnf, atnb, (size_t)S_ * LDO_);
  gemm_big(atnb, wob, out, S_, DIM_, LDO_);    // (2048x2048) -> d_out
}
